// EncodeProcessDecode_21990232556203
// MI455X (gfx1250) — matrix operand order repaired and checked by reading, not compiled, not run on hardware
//
#include <hip/hip_runtime.h>

#define HID 128
#define TILE_M 32

typedef __attribute__((ext_vector_type(16))) __bf16 v16bf;
typedef __attribute__((ext_vector_type(8)))  float  v8f;

union ABFrag {
    v16bf v;
    uint4 q[2];
    unsigned short us[16];
};

__device__ __forceinline__ unsigned short f2bf(float f) {
    unsigned int u = __float_as_uint(f);
    unsigned int r = u + 0x7FFFu + ((u >> 16) & 1u);   // round-to-nearest-even
    return (unsigned short)(r >> 16);
}

__device__ __forceinline__ float leakyf(float x) { return x > 0.f ? x : 0.01f * x; }

// ---------------------------------------------------------------------------
// Fused 2-layer MLP over 32-row tiles with gather/concat input.
//   X  = concat( s0[i0[r]] (k0), s1[i1[r]] (k1), s2[r] (k2) )   (bf16 in LDS)
//   Hd = leaky(X @ W1 + b1)      W1t: [128 col][KPAD k] bf16, zero-padded
//   Y  = Hd @ W2 + b2            W2t: [128 col][128  k] bf16, zero-padded
// mode 0: out[r*H2+n]  = Y
// mode 1: out[r*128+n] += Y                              (node residual)
// mode 2: aux[r*128+n] += Y; atomicAdd(scat[dst[r]*128+n], Y)  (edge msg)
// B fragments are read straight from global (weights live in L2).
// ---------------------------------------------------------------------------
template <int KPAD>
__global__ __launch_bounds__(256)
void mlp2_kernel(const float* __restrict__ s0, const int* __restrict__ i0, int k0,
                 const float* __restrict__ s1, const int* __restrict__ i1, int k1,
                 const float* __restrict__ s2, int k2,
                 const unsigned short* __restrict__ W1t, const float* __restrict__ B1,
                 const unsigned short* __restrict__ W2t, const float* __restrict__ B2,
                 int H2, int act2, int mode,
                 float* __restrict__ out, float* __restrict__ aux,
                 const int* __restrict__ dstIdx, float* __restrict__ scat,
                 int M)
{
    __shared__ unsigned short Xs[TILE_M * KPAD];
    __shared__ unsigned short Hs[TILE_M * HID];

    const int tid  = threadIdx.x;
    const int lane = tid & 31;
    const int wave = tid >> 5;
    const int row0 = blockIdx.x * TILE_M;

    const int Kin = k0 + k1 + k2;   // <= KPAD

    // ---- stage gathered/concatenated inputs as bf16 (zero padded) ----
    for (int e = tid; e < TILE_M * KPAD; e += 256) {
        int r = e / KPAD, k = e - r * KPAD;
        int grow = row0 + r;
        float val = 0.f;
        if (grow < M) {
            if (k < k0) {
                int sr = i0 ? i0[grow] : grow;
                val = s0[(long long)sr * k0 + k];
            } else if (k < k0 + k1) {
                int sr = i1 ? i1[grow] : grow;
                val = s1[(long long)sr * k1 + (k - k0)];
            } else if (k < Kin) {
                val = s2[(long long)grow * k2 + (k - k0 - k1)];
            }
        }
        Xs[r * KPAD + k] = f2bf(val);
    }
    __syncthreads();

    const int wr = (wave & 1) * 16;   // row group 0/16
    const int wc = (wave >> 1) * 32;  // col group 0/32/64/96 (two 16-wide tiles)

    const int asel = (lane >> 4) << 3;   // +0 / +8  (A-frag K start within tile)
    const int bsel = (lane >> 4) << 3;   // +0 / +8  (B-frag K start within tile)
    const int l15  = lane & 15;

    v8f c0, c1;
#pragma unroll
    for (int i = 0; i < 8; ++i) { c0[i] = 0.f; c1[i] = 0.f; }

    // ================= layer 1: KPAD -> 128, leaky =================
    {
        const unsigned short* wb0 = W1t + (size_t)(wc + l15) * KPAD + bsel;
        const unsigned short* wb1 = wb0 + (size_t)16 * KPAD;
        const unsigned short* pa0 = &Xs[(wr + l15) * KPAD + asel];
#pragma unroll
        for (int kt = 0; kt < KPAD; kt += 32) {
            ABFrag a, bf0, bf1;
            a.q[0] = *(const uint4*)(pa0 + kt);
            a.q[1] = *(const uint4*)(pa0 + kt + 16);
            bf0.q[0] = *(const uint4*)(wb0 + kt);
            bf0.q[1] = *(const uint4*)(wb0 + kt + 16);
            bf1.q[0] = *(const uint4*)(wb1 + kt);
            bf1.q[1] = *(const uint4*)(wb1 + kt + 16);
            c0 = __builtin_amdgcn_wmma_f32_16x16x32_bf16(false, a.v, false, bf0.v, (short)0, c0, false, false);
            c1 = __builtin_amdgcn_wmma_f32_16x16x32_bf16(false, a.v, false, bf1.v, (short)0, c1, false, false);
        }
    }

    // epilogue 1: bias + leaky -> Hs (bf16)
    {
        int mrow = wr + ((lane < 16) ? 0 : 8);
#pragma unroll
        for (int v = 0; v < 8; ++v) {
            int m  = mrow + v;
            int n0 = wc + l15;
            int n1 = wc + 16 + l15;
            Hs[m * HID + n0] = f2bf(leakyf(c0[v] + B1[n0]));
            Hs[m * HID + n1] = f2bf(leakyf(c1[v] + B1[n1]));
        }
    }
    __syncthreads();

#pragma unroll
    for (int i = 0; i < 8; ++i) { c0[i] = 0.f; c1[i] = 0.f; }

    // ================= layer 2: 128 -> H2 =================
    {
        const unsigned short* wb0 = W2t + (size_t)(wc + l15) * HID + bsel;
        const unsigned short* wb1 = wb0 + (size_t)16 * HID;
        const unsigned short* pa0 = &Hs[(wr + l15) * HID + asel];
#pragma unroll
        for (int kt = 0; kt < HID; kt += 32) {
            ABFrag a, bf0, bf1;
            a.q[0] = *(const uint4*)(pa0 + kt);
            a.q[1] = *(const uint4*)(pa0 + kt + 16);
            bf0.q[0] = *(const uint4*)(wb0 + kt);
            bf0.q[1] = *(const uint4*)(wb0 + kt + 16);
            bf1.q[0] = *(const uint4*)(wb1 + kt);
            bf1.q[1] = *(const uint4*)(wb1 + kt + 16);
            c0 = __builtin_amdgcn_wmma_f32_16x16x32_bf16(false, a.v, false, bf0.v, (short)0, c0, false, false);
            c1 = __builtin_amdgcn_wmma_f32_16x16x32_bf16(false, a.v, false, bf1.v, (short)0, c1, false, false);
        }
    }

    // epilogue 2
    {
        int mrow = wr + ((lane < 16) ? 0 : 8);
#pragma unroll
        for (int v = 0; v < 8; ++v) {
            int m = mrow + v;
            int grow = row0 + m;
            if (grow >= M) continue;
#pragma unroll
            for (int t = 0; t < 2; ++t) {
                int n = wc + t * 16 + l15;
                if (n >= H2) continue;
                float val = (t == 0 ? c0[v] : c1[v]) + B2[n];
                if (act2) val = leakyf(val);
                if (mode == 0) {
                    out[(long long)grow * H2 + n] = val;
                } else if (mode == 1) {
                    out[(long long)grow * HID + n] += val;
                } else {
                    aux[(long long)grow * HID + n] += val;
                    atomicAdd(&scat[(long long)dstIdx[grow] * HID + n], val);
                }
            }
        }
    }
}

// ---------------------------------------------------------------------------
// Convert f32 weight [Kin x Ncols] -> bf16 transposed padded [128 x KPAD]
__global__ void cvtT_kernel(const float* __restrict__ src, unsigned short* __restrict__ dst,
                            int Kin, int Ncols, int KPAD) {
    int i = blockIdx.x * 256 + threadIdx.x;
    if (i >= HID * KPAD) return;
    int col = i / KPAD, k = i - col * KPAD;
    unsigned short v = 0;
    if (k < Kin && col < Ncols) v = f2bf(src[(long long)k * Ncols + col]);
    dst[i] = v;
}

__global__ void deg_kernel(const int* __restrict__ dst, float* __restrict__ deg, int E) {
    int i = blockIdx.x * 256 + threadIdx.x;
    if (i < E) atomicAdd(&deg[dst[i]], 1.0f);
}

__global__ void aggr_kernel(float* __restrict__ s, const float* __restrict__ deg, int n) {
    int i = blockIdx.x * 256 + threadIdx.x;
    if (i < n) {
        int r = i / HID;
        s[i] = s[i] / fmaxf(deg[r], 1.0f);
    }
}

__global__ __launch_bounds__(256)
void colmax_kernel(const float* __restrict__ nf, const float* __restrict__ tptr,
                   float* __restrict__ cmax, int n) {
    __shared__ float red[256];
    int c = blockIdx.x;
    float t = *tptr;
    float m = -3.402823466e38f;
    for (int i = threadIdx.x; i < n; i += 256) m = fmaxf(m, t * nf[(long long)i * HID + c]);
    red[threadIdx.x] = m;
    __syncthreads();
    for (int s = 128; s > 0; s >>= 1) {
        if (threadIdx.x < s) red[threadIdx.x] = fmaxf(red[threadIdx.x], red[threadIdx.x + s]);
        __syncthreads();
    }
    if (threadIdx.x == 0) cmax[c] = red[0];
}

__global__ __launch_bounds__(256)
void colsum_kernel(const float* __restrict__ nf, const float* __restrict__ tptr,
                   const float* __restrict__ cmax, float* __restrict__ glob, int n) {
    __shared__ float rd[256], rn[256];
    int c = blockIdx.x;
    float t = *tptr, m = cmax[c];
    float sd = 0.f, sn = 0.f;
    for (int i = threadIdx.x; i < n; i += 256) {
        float x = nf[(long long)i * HID + c];
        float z = __expf(t * x - m);
        sd += z;
        sn += z * x;
    }
    rd[threadIdx.x] = sd; rn[threadIdx.x] = sn;
    __syncthreads();
    for (int s = 128; s > 0; s >>= 1) {
        if (threadIdx.x < s) { rd[threadIdx.x] += rd[threadIdx.x + s]; rn[threadIdx.x] += rn[threadIdx.x + s]; }
        __syncthreads();
    }
    if (threadIdx.x == 0) glob[c] = rn[0] / rd[0];
}

__global__ __launch_bounds__(128)
void decg_kernel(const float* __restrict__ glob,
                 const float* __restrict__ W1, const float* __restrict__ B1,
                 const float* __restrict__ W2, const float* __restrict__ B2,
                 const float* __restrict__ W3, const float* __restrict__ B3,
                 float* __restrict__ outg) {
    __shared__ float g[HID], h1[HID], h2[HID];
    int j = threadIdx.x;
    g[j] = glob[j];
    __syncthreads();
    float acc = B1[j];
    for (int k = 0; k < HID; ++k) acc += g[k] * W1[k * HID + j];
    h1[j] = leakyf(acc);
    __syncthreads();
    acc = B2[j];
    for (int k = 0; k < HID; ++k) acc += h1[k] * W2[k * HID + j];
    h2[j] = leakyf(acc);
    __syncthreads();
    if (j < 2) {
        float a = B3[j];
        for (int k = 0; k < HID; ++k) a += h2[k] * W3[k * 2 + j];
        outg[j] = a;
    }
}

// ---------------------------------------------------------------------------
extern "C" void kernel_launch(void* const* d_in, const int* in_sizes, int n_in,
                              void* d_out, int out_size, void* d_ws, size_t ws_size,
                              hipStream_t stream) {
    const int N = 100000, E = 400000, NF = 8, EF = 4, NB = 5;

    const float* xin = (const float*)d_in[0];
    const int*   ei  = (const int*)d_in[1];
    const float* ea  = (const float*)d_in[2];
    const int* srcI = ei;       // edge_index[0] = x_j source
    const int* dstI = ei + E;   // edge_index[1] = x_i target

    auto P = [&](int i) { return (const float*)d_in[4 + i]; };
    // params tree order: blocks[5]{edge(W1,b1,W2,b2), node(W1,b1,W2,b2)},
    //                    dec_g(W1,b1,W2,b2,W3,b3), dec_n(W1,b1,W2,b2),
    //                    enc_e(W1,b1,W2,b2), enc_n(W1,b1,W2,b2), t
    const float* decg_W1 = P(40); const float* decg_b1 = P(41);
    const float* decg_W2 = P(42); const float* decg_b2 = P(43);
    const float* decg_W3 = P(44); const float* decg_b3 = P(45);
    const float* decn_W1 = P(46); const float* decn_b1 = P(47);
    const float* decn_W2 = P(48); const float* decn_b2 = P(49);
    const float* ence_W1 = P(50); const float* ence_b1 = P(51);
    const float* ence_W2 = P(52); const float* ence_b2 = P(53);
    const float* encn_W1 = P(54); const float* encn_b1 = P(55);
    const float* encn_W2 = P(56); const float* encn_b2 = P(57);
    const float* tptr    = P(58);

    // ---- workspace carve ----
    char* wsb = (char*)d_ws;
    size_t off = 0;
    auto carve = [&](size_t bytes) -> void* {
        void* p = wsb + off;
        off = (off + bytes + 255) & ~(size_t)255;
        return p;
    };
    float* nf   = (float*)carve((size_t)N * HID * 4);
    float* ef   = (float*)carve((size_t)E * HID * 4);
    float* sbuf = (float*)carve((size_t)N * HID * 4);
    float* deg  = (float*)carve((size_t)N * 4);
    float* cmax = (float*)carve(HID * 4);
    float* glob = (float*)carve(HID * 4);

    // transposed, zero-padded bf16 weights: [128 cols][KPAD]
    unsigned short* b_encn1 = (unsigned short*)carve((size_t)HID * 32 * 2);
    unsigned short* b_encn2 = (unsigned short*)carve((size_t)HID * HID * 2);
    unsigned short* b_ence1 = (unsigned short*)carve((size_t)HID * 32 * 2);
    unsigned short* b_ence2 = (unsigned short*)carve((size_t)HID * HID * 2);
    unsigned short* b_decn1 = (unsigned short*)carve((size_t)HID * HID * 2);
    unsigned short* b_decn2 = (unsigned short*)carve((size_t)HID * HID * 2);
    unsigned short *b_eW1[NB], *b_eW2[NB], *b_nW1[NB], *b_nW2[NB];
    for (int b = 0; b < NB; ++b) {
        b_eW1[b] = (unsigned short*)carve((size_t)HID * 384 * 2);
        b_eW2[b] = (unsigned short*)carve((size_t)HID * HID * 2);
        b_nW1[b] = (unsigned short*)carve((size_t)HID * 256 * 2);
        b_nW2[b] = (unsigned short*)carve((size_t)HID * HID * 2);
    }

    auto cvtT = [&](const float* w, unsigned short* o, int Kin, int Ncols, int KPAD) {
        int n = HID * KPAD;
        cvtT_kernel<<<(n + 255) / 256, 256, 0, stream>>>(w, o, Kin, Ncols, KPAD);
    };
    cvtT(encn_W1, b_encn1, NF, HID, 32);
    cvtT(encn_W2, b_encn2, HID, HID, HID);
    cvtT(ence_W1, b_ence1, EF, HID, 32);
    cvtT(ence_W2, b_ence2, HID, HID, HID);
    cvtT(decn_W1, b_decn1, HID, HID, HID);
    cvtT(decn_W2, b_decn2, HID, 3, HID);
    for (int b = 0; b < NB; ++b) {
        cvtT(P(b * 8 + 0), b_eW1[b], 3 * HID, HID, 384);
        cvtT(P(b * 8 + 2), b_eW2[b], HID, HID, HID);
        cvtT(P(b * 8 + 4), b_nW1[b], 2 * HID, HID, 256);
        cvtT(P(b * 8 + 6), b_nW2[b], HID, HID, HID);
    }

    // degree (constant across blocks)
    hipMemsetAsync(deg, 0, (size_t)N * 4, stream);
    deg_kernel<<<(E + 255) / 256, 256, 0, stream>>>(dstI, deg, E);

    const int gridN = (N + TILE_M - 1) / TILE_M;
    const int gridE = (E + TILE_M - 1) / TILE_M;

    // encoders
    mlp2_kernel<32><<<gridN, 256, 0, stream>>>(xin, nullptr, NF, nullptr, nullptr, 0, nullptr, 0,
                                               b_encn1, encn_b1, b_encn2, encn_b2,
                                               HID, 0, 0, nf, nullptr, nullptr, nullptr, N);
    mlp2_kernel<32><<<gridE, 256, 0, stream>>>(ea, nullptr, EF, nullptr, nullptr, 0, nullptr, 0,
                                               b_ence1, ence_b1, b_ence2, ence_b2,
                                               HID, 0, 0, ef, nullptr, nullptr, nullptr, E);

    // message passing blocks
    for (int b = 0; b < NB; ++b) {
        const float* eb1 = P(b * 8 + 1); const float* eb2 = P(b * 8 + 3);
        const float* nb1 = P(b * 8 + 5); const float* nb2 = P(b * 8 + 7);

        hipMemsetAsync(sbuf, 0, (size_t)N * HID * 4, stream);
        // msg = edgeMLP([nf[dst], nf[src], ef]); ef += msg; s[dst] += msg
        mlp2_kernel<384><<<gridE, 256, 0, stream>>>(nf, dstI, HID, nf, srcI, HID, ef, HID,
                                                    b_eW1[b], eb1, b_eW2[b], eb2,
                                                    HID, 0, 2, nullptr, ef, dstI, sbuf, E);
        // aggr = s / max(deg,1)
        aggr_kernel<<<(N * HID + 255) / 256, 256, 0, stream>>>(sbuf, deg, N * HID);
        // nf += nodeMLP([nf, aggr])
        mlp2_kernel<256><<<gridN, 256, 0, stream>>>(nf, nullptr, HID, sbuf, nullptr, HID, nullptr, 0,
                                                    b_nW1[b], nb1, b_nW2[b], nb2,
                                                    HID, 0, 1, nf, nullptr, nullptr, nullptr, N);
    }

    // node decoder -> y  [N,3]
    mlp2_kernel<128><<<gridN, 256, 0, stream>>>(nf, nullptr, HID, nullptr, nullptr, 0, nullptr, 0,
                                                b_decn1, decn_b1, b_decn2, decn_b2,
                                                3, 0, 0, (float*)d_out, nullptr, nullptr, nullptr, N);

    // global softmax aggregation + graph decoder -> d_out[N*3 .. N*3+2)
    colmax_kernel<<<HID, 256, 0, stream>>>(nf, tptr, cmax, N);
    colsum_kernel<<<HID, 256, 0, stream>>>(nf, tptr, cmax, glob, N);
    decg_kernel<<<1, 128, 0, stream>>>(glob, decg_W1, decg_b1, decg_W2, decg_b2,
                                       decg_W3, decg_b3, (float*)d_out + (size_t)N * 3);
}
